// TransformerModel_60885456388608
// MI455X (gfx1250) — compile-verified, matrix operand order checked by reading, not run on hardware
//
#include <hip/hip_runtime.h>
#include <hip/hip_bf16.h>

// ---------------------------------------------------------------------------
// Model constants (match reference)
// ---------------------------------------------------------------------------
constexpr int TB    = 8;       // batch
constexpr int TS    = 512;     // seq len (src == tgt)
constexpr int TD    = 1024;    // model dim
constexpr int TH    = 16;      // heads
constexpr int THD   = 64;      // head dim
constexpr int TFF   = 4096;    // ffn dim
constexpr int TV    = 16000;   // vocab
constexpr int TOK   = TB * TS; // 4096 tokens

typedef __attribute__((ext_vector_type(16))) __bf16 v16bf;
typedef __attribute__((ext_vector_type(8)))  __bf16 v8bf;
typedef __attribute__((ext_vector_type(4)))  __bf16 v4bf;
typedef __attribute__((ext_vector_type(8)))  float  v8f;

// ---------------------------------------------------------------------------
// CDNA5 async global->LDS copy (ASYNCcnt-tracked DMA, bypasses VGPRs)
// ---------------------------------------------------------------------------
__device__ __forceinline__ void async_copy16(unsigned lds_addr, const void* gaddr) {
    asm volatile("global_load_async_to_lds_b128 %0, %1, off"
                 :: "v"(lds_addr), "v"(gaddr) : "memory");
}
__device__ __forceinline__ void async_wait0() {
    asm volatile("s_wait_asynccnt 0x0" ::: "memory");
}

// ---------------------------------------------------------------------------
// WMMA GEMM:  C[M,N] = alpha * A[M,K] x Bw[N,K]^T (+ bias, + relu)
//   A, Bw are bf16 (pre-converted). Requires M%128==0, K%32==0 (guaranteed by
//   caller); N may be ragged (B-row index clamped, store guarded).
//   Batched via blockIdx.z: off = (z/Hdiv)*s? + (z%Hdiv)*t?
// Block: 256 thr (8 waves, 2Mx4N). Tile 128x256x32, wave tile 64x64
// (4x4 WMMA = 16 wmma per 16 ds_load_b128), double-buffered async-DMA LDS.
// ---------------------------------------------------------------------------
#define BM 128
#define BN 256
#define BK 32
#define LSTR 40   // LDS row stride in bf16 elems (80B: 16B-aligned, bank-spread)

__device__ __forceinline__ v16bf load_frag(const __bf16* p, int row, int klo) {
    const __bf16* q = p + row * LSTR;
    v8bf lo = *(const v8bf*)(q + klo);
    v8bf hi = *(const v8bf*)(q + klo + 16);
    return __builtin_shufflevector(lo, hi, 0,1,2,3,4,5,6,7,8,9,10,11,12,13,14,15);
}

__global__ __launch_bounds__(256)
void gemm_bf16_wmma_kernel(const __bf16* __restrict__ A, const __bf16* __restrict__ Bw,
                           const float* __restrict__ bias, float* __restrict__ C,
                           int M, int N, int K, int lda, int ldb, int ldc,
                           int Hdiv,
                           long long sA, long long tA,
                           long long sB, long long tB,
                           long long sC, long long tC,
                           float alpha, int relu)
{
    __shared__ __bf16 As[2][BM * LSTR];
    __shared__ __bf16 Bs[2][BN * LSTR];

    const int z  = blockIdx.z;
    const int bb = z / Hdiv, hh = z % Hdiv;
    const long long offB = (long long)bb * sB + (long long)hh * tB;
    const long long offC = (long long)bb * sC + (long long)hh * tC;

    const int tid  = threadIdx.x;
    const int lane = tid & 31;
    const int wave = tid >> 5;
    const int wm   = (wave >> 2) * 64;   // 0,64
    const int wn   = (wave & 3) * 64;    // 0,64,128,192
    const int lr   = lane & 15;          // row/col within 16x16 tile
    const int hig  = lane >> 4;          // lane group 0/1
    const int klo  = hig * 8;            // K start for this lane's halves

    const int blockN = blockIdx.x * BN;
    const int blockM = blockIdx.y * BM;

    // staging: A tile 128x32 = 512 x 16B chunks (2/thr), B tile 256x32 = 1024 (4/thr)
    const int rw = tid >> 2;        // 0..63
    const int pt = (tid & 3) * 8;   // K sub-offset in elements (0,8,16,24)
    const __bf16* Abase = A + ((long long)bb * sA + (long long)hh * tA)
                            + (long long)blockM * lda;
    int rb0 = blockN + rw;       if (rb0 >= N) rb0 = N - 1;   // clamp ragged N
    int rb1 = blockN + rw + 64;  if (rb1 >= N) rb1 = N - 1;
    int rb2 = blockN + rw + 128; if (rb2 >= N) rb2 = N - 1;
    int rb3 = blockN + rw + 192; if (rb3 >= N) rb3 = N - 1;

    v8f acc[4][4] = {};

    auto stage = [&](int buf, int kt) {
        async_copy16((unsigned)(size_t)&As[buf][rw * LSTR + pt],
                     Abase + (long long)rw * lda + kt + pt);
        async_copy16((unsigned)(size_t)&As[buf][(rw + 64) * LSTR + pt],
                     Abase + (long long)(rw + 64) * lda + kt + pt);
        async_copy16((unsigned)(size_t)&Bs[buf][rw * LSTR + pt],
                     Bw + offB + (long long)rb0 * ldb + kt + pt);
        async_copy16((unsigned)(size_t)&Bs[buf][(rw + 64) * LSTR + pt],
                     Bw + offB + (long long)rb1 * ldb + kt + pt);
        async_copy16((unsigned)(size_t)&Bs[buf][(rw + 128) * LSTR + pt],
                     Bw + offB + (long long)rb2 * ldb + kt + pt);
        async_copy16((unsigned)(size_t)&Bs[buf][(rw + 192) * LSTR + pt],
                     Bw + offB + (long long)rb3 * ldb + kt + pt);
    };

    const int ktiles = K / BK;
    stage(0, 0);
    for (int t = 0; t < ktiles; ++t) {
        const int buf = t & 1;
        async_wait0();       // my DMA chunks for this buffer have landed
        __syncthreads();     // everyone's have, and prior reads are done
        if (t + 1 < ktiles) stage(buf ^ 1, (t + 1) * BK);  // overlap with math

        v16bf af[4], bfr[4];
        #pragma unroll
        for (int m = 0; m < 4; ++m) af[m]  = load_frag(As[buf], wm + m * 16 + lr, klo);
        #pragma unroll
        for (int n = 0; n < 4; ++n) bfr[n] = load_frag(Bs[buf], wn + n * 16 + lr, klo);

        #pragma unroll
        for (int m = 0; m < 4; ++m)
            #pragma unroll
            for (int n = 0; n < 4; ++n)
                acc[m][n] = __builtin_amdgcn_wmma_f32_16x16x32_bf16(
                    false, af[m], false, bfr[n], (short)0, acc[m][n], false, false);
    }

    // epilogue (C/D layout: VGPR r -> M = r + hig*8, lane -> N)
    #pragma unroll
    for (int n = 0; n < 4; ++n) {
        int gcol = blockN + wn + n * 16 + lr;
        int bcol = gcol < N ? gcol : N - 1;
        float bv = (bias != nullptr) ? bias[bcol] : 0.f;
        #pragma unroll
        for (int m = 0; m < 4; ++m) {
            #pragma unroll
            for (int r = 0; r < 8; ++r) {
                int grow = blockM + wm + m * 16 + hig * 8 + r;
                if (grow < M && gcol < N) {
                    float v = acc[m][n][r] * alpha + bv;
                    if (relu) v = fmaxf(v, 0.f);
                    C[offC + (long long)grow * ldc + gcol] = v;
                }
            }
        }
    }
}

// ---------------------------------------------------------------------------
// fp32 -> bf16 bulk convert (vectorized). n % 4 == 0.
// ---------------------------------------------------------------------------
__global__ __launch_bounds__(256)
void cvt_f32_bf16_kernel(const float* __restrict__ src, __bf16* __restrict__ dst,
                         long long n)
{
    long long i = ((long long)blockIdx.x * 256 + threadIdx.x) * 4;
    if (i >= n) return;
    float4 v = *(const float4*)(src + i);
    v4bf o;
    o[0] = (__bf16)v.x; o[1] = (__bf16)v.y; o[2] = (__bf16)v.z; o[3] = (__bf16)v.w;
    *(v4bf*)(dst + i) = o;
}

// ---------------------------------------------------------------------------
// [B,S,H,HD] fp32 -> per-head bf16: tr==0 -> [B*H,S,HD], tr==1 -> [B*H,HD,S]
// ---------------------------------------------------------------------------
__global__ __launch_bounds__(256)
void heads_cvt_kernel(const float* __restrict__ src, __bf16* __restrict__ dst, int tr)
{
    int e = blockIdx.x * 256 + threadIdx.x;       // < TOK*TD
    int d = e % THD;
    int h = (e / THD) % TH;
    int s = (e / (THD * TH)) % TS;
    int b = e / (THD * TH * TS);
    int bh = b * TH + h;
    long long di = tr ? ((long long)bh * THD + d) * TS + s
                      : ((long long)bh * TS + s) * THD + d;
    dst[di] = (__bf16)src[e];
}

// ---------------------------------------------------------------------------
// Row softmax over scores [B*H*Sq, Sk=512]; one wave32 per row.
// ---------------------------------------------------------------------------
__global__ __launch_bounds__(256)
void softmax_kernel(float* __restrict__ S, int Sq, int totalRows, int causal)
{
    int row  = blockIdx.x * 8 + (threadIdx.x >> 5);
    int lane = threadIdx.x & 31;
    if (row >= totalRows) return;
    long long base = (long long)row * TS;
    int qpos = row % Sq;

    float vals[16];
    float mx = -3.4e38f;
    #pragma unroll
    for (int i = 0; i < 16; ++i) {
        int col = lane + i * 32;
        float v = S[base + col];
        if (causal && col > qpos) v = -3.4e38f;
        vals[i] = v;
        mx = fmaxf(mx, v);
    }
    #pragma unroll
    for (int off = 16; off > 0; off >>= 1) mx = fmaxf(mx, __shfl_xor(mx, off));

    float sum = 0.f;
    #pragma unroll
    for (int i = 0; i < 16; ++i) {
        int col = lane + i * 32;
        float p = (causal && col > qpos) ? 0.f : __expf(vals[i] - mx);
        vals[i] = p;
        sum += p;
    }
    #pragma unroll
    for (int off = 16; off > 0; off >>= 1) sum += __shfl_xor(sum, off);
    float inv = 1.f / sum;
    #pragma unroll
    for (int i = 0; i < 16; ++i) S[base + lane + i * 32] = vals[i] * inv;
}

// ---------------------------------------------------------------------------
// out = LayerNorm(x + r) * g + b    (row = one block of 256, D = 1024)
// ---------------------------------------------------------------------------
__global__ __launch_bounds__(256)
void add_ln_kernel(const float* __restrict__ x, const float* __restrict__ r,
                   const float* __restrict__ g, const float* __restrict__ b,
                   float* __restrict__ out)
{
    __shared__ float sh1[8], sh2[8];
    long long base = (long long)blockIdx.x * TD;
    int tid = threadIdx.x, lane = tid & 31, wave = tid >> 5;

    float loc[4], s1 = 0.f, s2 = 0.f;
    #pragma unroll
    for (int i = 0; i < 4; ++i) {
        int d = tid + i * 256;
        float v = x[base + d] + r[base + d];
        loc[i] = v; s1 += v; s2 += v * v;
    }
    #pragma unroll
    for (int off = 16; off > 0; off >>= 1) { s1 += __shfl_xor(s1, off); s2 += __shfl_xor(s2, off); }
    if (lane == 0) { sh1[wave] = s1; sh2[wave] = s2; }
    __syncthreads();
    s1 = 0.f; s2 = 0.f;
    #pragma unroll
    for (int w = 0; w < 8; ++w) { s1 += sh1[w]; s2 += sh2[w]; }
    float mu = s1 / TD;
    float var = s2 / TD - mu * mu;
    float rinv = rsqrtf(var + 1e-5f);
    #pragma unroll
    for (int i = 0; i < 4; ++i) {
        int d = tid + i * 256;
        out[base + d] = (loc[i] - mu) * rinv * g[d] + b[d];
    }
}

// ---------------------------------------------------------------------------
// out[t, :] = emb[tok[t], :] + sinusoidal PE(pos = t % S)
// ---------------------------------------------------------------------------
__global__ __launch_bounds__(256)
void embed_kernel(const int* __restrict__ tok, const float* __restrict__ emb,
                  float* __restrict__ out)
{
    int t = blockIdx.x;
    int pos = t % TS;
    int id = tok[t];
    #pragma unroll
    for (int i = 0; i < 4; ++i) {
        int d = threadIdx.x + i * 256;
        float div = __expf((float)(d & ~1) * (-9.210340371976184f / (float)TD));
        float ang = (float)pos * div;
        float pe = (d & 1) ? __cosf(ang) : __sinf(ang);
        out[(long long)t * TD + d] = emb[(long long)id * TD + d] + pe;
    }
}

// ---------------------------------------------------------------------------
// Host side
// ---------------------------------------------------------------------------
static void launch_gemm(hipStream_t stream,
    const __bf16* A, const __bf16* Bw, const float* bias, float* C,
    int M, int N, int K, int lda, int ldb, int ldc,
    int batches, int Hdiv,
    long long sA, long long tA, long long sB, long long tB,
    long long sC, long long tC,
    float alpha, int relu)
{
    dim3 grid((N + BN - 1) / BN, (M + BM - 1) / BM, batches);
    gemm_bf16_wmma_kernel<<<grid, dim3(256), 0, stream>>>(
        A, Bw, bias, C, M, N, K, lda, ldb, ldc, Hdiv,
        sA, tA, sB, tB, sC, tC, alpha, relu);
}

extern "C" void kernel_launch(void* const* d_in, const int* in_sizes, int n_in,
                              void* d_out, int out_size, void* d_ws, size_t ws_size,
                              hipStream_t stream) {
    (void)in_sizes; (void)n_in; (void)out_size; (void)ws_size;

    // ---- parameter leaf indices (jax tree-flatten order: sorted dict keys) ----
    // top: params, src, tgt.  params: dec(4x26), enc(4x16), out(2), src_emb, tgt_emb
    auto P = [&](int i) { return (const float*)d_in[i]; };
    auto decBase = [](int l) { return l * 26; };
    auto encBase = [](int l) { return 104 + l * 16; };
    const int OUT_B = 168, OUT_W = 169, SRC_EMB = 170, TGT_EMB = 171, SRC = 172, TGT = 173;

    // ---- workspace layout ----
    char* wsb = (char*)d_ws;
    size_t off = 0;
    auto alloc = [&](size_t bytes) {
        void* p = wsb + off; off += (bytes + 255) & ~(size_t)255; return p;
    };
    // fp32 buffers
    float* A0   = (float*)alloc((size_t)TOK * TD * 4);
    float* A1   = (float*)alloc((size_t)TOK * TD * 4);
    float* Qf   = (float*)alloc((size_t)TOK * TD * 4);
    float* Kf   = (float*)alloc((size_t)TOK * TD * 4);
    float* Vf   = (float*)alloc((size_t)TOK * TD * 4);
    float* CXf  = (float*)alloc((size_t)TOK * TD * 4);
    float* MEM  = (float*)alloc((size_t)TOK * TD * 4);
    float* FFH  = (float*)alloc((size_t)TOK * TFF * 4);
    float* SC   = (float*)alloc((size_t)TB * TH * TS * TS * 4);
    // bf16 buffers
    __bf16* Abf  = (__bf16*)alloc((size_t)TOK * TFF * 2);        // largest A (M*K)
    __bf16* A2bf = (__bf16*)alloc((size_t)TOK * TD * 2);         // cross-attn kv
    __bf16* Wbf  = (__bf16*)alloc((size_t)TV * TD * 2);          // largest weight
    __bf16* Qbf  = (__bf16*)alloc((size_t)TOK * TD * 2);         // [B*H,S,HD]
    __bf16* Kbf  = (__bf16*)alloc((size_t)TOK * TD * 2);         // [B*H,S,HD]
    __bf16* Vtbf = (__bf16*)alloc((size_t)TOK * TD * 2);         // [B*H,HD,S]
    __bf16* SCbf = (__bf16*)alloc((size_t)TB * TH * TS * TS * 2);

    const float iscale = 0.125f; // 1/sqrt(HD)
    const long long SSQ = (long long)TS * TS;

    auto cvt = [&](const float* s, __bf16* d, long long n) {
        cvt_f32_bf16_kernel<<<dim3((unsigned)((n / 4 + 255) / 256)), dim3(256), 0, stream>>>(s, d, n);
    };
    auto headcvt = [&](const float* s, __bf16* d, int tr) {
        heads_cvt_kernel<<<dim3(TOK * TD / 256), dim3(256), 0, stream>>>(s, d, tr);
    };
    // plain (non-batched) linear: C[M,N] = x(bf16) * W^T + bias
    auto lin = [&](const __bf16* xbf, int wi, int bi, float* C, int M, int N, int K,
                   int relu) {
        cvt(P(wi), Wbf, (long long)N * K);
        launch_gemm(stream, xbf, Wbf, P(bi), C, M, N, K, K, K, N,
                    1, 1, 0, 0, 0, 0, 0, 0, 1.f, relu);
    };

    // ---- multi-head attention ----
    auto mha = [&](const float* xq, const float* xkv,
                   int qw, int qb, int kw, int kb, int vw, int vb, int ow, int ob,
                   int causal, float* outBuf) {
        cvt(xq, Abf, (long long)TOK * TD);
        const __bf16* kvbf = Abf;
        if (xkv != xq) { cvt(xkv, A2bf, (long long)TOK * TD); kvbf = A2bf; }
        lin(Abf,  qw, qb, Qf, TOK, TD, TD, 0);
        lin(kvbf, kw, kb, Kf, TOK, TD, TD, 0);
        lin(kvbf, vw, vb, Vf, TOK, TD, TD, 0);
        headcvt(Qf, Qbf, 0);
        headcvt(Kf, Kbf, 0);
        headcvt(Vf, Vtbf, 1);
        // scores[z] = Q[z] (SxHD) x K[z]^T, scaled
        launch_gemm(stream, Qbf, Kbf, nullptr, SC, TS, TS, THD, THD, THD, TS,
                    TB * TH, 1,
                    (long long)TS * THD, 0,
                    (long long)TS * THD, 0,
                    SSQ, 0,
                    iscale, 0);
        softmax_kernel<<<dim3((TB * TH * TS) / 8), dim3(256), 0, stream>>>(
            SC, TS, TB * TH * TS, causal);
        cvt(SC, SCbf, (long long)TB * TH * SSQ);
        // ctx[z] = attn (SxS) x Vt[z]^T (Vt is [HD,S] -> weight form [N=HD,K=S])
        launch_gemm(stream, SCbf, Vtbf, nullptr, CXf, TS, THD, TS, TS, TS, TD,
                    TB * TH, TH,
                    (long long)TH * SSQ, SSQ,
                    (long long)TH * THD * TS, (long long)THD * TS,
                    (long long)TS * TD, THD,
                    1.f, 0);
        cvt(CXf, Abf, (long long)TOK * TD);
        lin(Abf, ow, ob, outBuf, TOK, TD, TD, 0);
    };

    auto ffn = [&](const float* x, int w1, int b1, int w2, int b2, float* outBuf) {
        cvt(x, Abf, (long long)TOK * TD);
        lin(Abf, w1, b1, FFH, TOK, TFF, TD, 1);
        cvt(FFH, Abf, (long long)TOK * TFF);
        lin(Abf, w2, b2, outBuf, TOK, TD, TFF, 0);
    };

    auto add_ln = [&](const float* x, const float* r, int lb, int lg, float* outBuf) {
        add_ln_kernel<<<dim3(TOK), dim3(256), 0, stream>>>(x, r, P(lg), P(lb), outBuf);
    };

    // ================= encoder =================
    embed_kernel<<<dim3(TOK), dim3(256), 0, stream>>>((const int*)d_in[SRC], P(SRC_EMB), A0);
    for (int l = 0; l < 4; ++l) {
        int e = encBase(l);
        mha(A0, A0, e + 13, e + 12, e + 9, e + 8, e + 15, e + 14, e + 11, e + 10, 0, A1);
        add_ln(A0, A1, e + 4, e + 5, A0);                       // ln1
        ffn(A0, e + 1, e + 0, e + 3, e + 2, A1);                // ff1, ff2
        add_ln(A0, A1, e + 6, e + 7, A0);                       // ln2
    }
    hipMemcpyAsync(MEM, A0, (size_t)TOK * TD * 4, hipMemcpyDeviceToDevice, stream);

    // ================= decoder =================
    embed_kernel<<<dim3(TOK), dim3(256), 0, stream>>>((const int*)d_in[TGT], P(TGT_EMB), A0);
    for (int l = 0; l < 4; ++l) {
        int d = decBase(l);
        mha(A0, A0, d + 23, d + 22, d + 19, d + 18, d + 25, d + 24, d + 21, d + 20, 1, A1);
        add_ln(A0, A1, d + 12, d + 13, A0);                     // ln1
        mha(A0, MEM, d + 5, d + 4, d + 1, d + 0, d + 7, d + 6, d + 3, d + 2, 0, A1);
        add_ln(A0, A1, d + 14, d + 15, A0);                     // ln2
        ffn(A0, d + 9, d + 8, d + 11, d + 10, A1);              // ff1, ff2
        add_ln(A0, A1, d + 16, d + 17, A0);                     // ln3
    }

    // ================= output projection -> logits =================
    cvt(A0, Abf, (long long)TOK * TD);
    lin(Abf, OUT_W, OUT_B, (float*)d_out, TOK, TV, TD, 0);
}
